// CausalTemporalReasoningAdapter_74105365725293
// MI455X (gfx1250) — matrix operand order repaired and checked by reading, not compiled, not run on hardware
//
#include <hip/hip_runtime.h>
#include <hip/hip_bf16.h>
#include <math.h>

typedef __attribute__((ext_vector_type(16))) _Float16 v16h;
typedef __attribute__((ext_vector_type(8)))  float    v8f;

#define NV 24
#define NB 8
#define NS 512
#define NROWS (NB*NS)        // 4096
#define NH1 128
#define NH2 64
#define NSH1 256
#define NSH2 128
#define NIH 32
#define NP (NV*NV)           // 576

// workspace layout (floats)
#define WS_CORR 0
#define WS_ADJ  576
#define WS_HP   1152

// ---------------------------------------------------------------------------
// Kernel 1: batch-average, covariance, correlation -> ws[WS_CORR..]
// ---------------------------------------------------------------------------
__global__ void __launch_bounds__(256)
k_stats(const float* __restrict__ data, float* __restrict__ ws) {
    __shared__ float s_avg[NS * NV];   // 48 KB
    __shared__ float s_mean[NV];
    __shared__ float s_cov[NV * NV];
    int t = threadIdx.x;

    for (int e = t; e < NS * NV; e += 256) {
        float a = 0.f;
        #pragma unroll
        for (int b = 0; b < NB; ++b) a += data[b * (NS * NV) + e];
        s_avg[e] = a * (1.0f / NB);
    }
    __syncthreads();
    if (t < NV) {
        float m = 0.f;
        for (int s = 0; s < NS; ++s) m += s_avg[s * NV + t];
        s_mean[t] = m * (1.0f / NS);
    }
    __syncthreads();
    for (int e = t; e < NV * NV; e += 256) {
        int i = e / NV, j = e % NV;
        float mi = s_mean[i], mj = s_mean[j];
        float acc = 0.f;
        for (int s = 0; s < NS; ++s)
            acc += (s_avg[s * NV + i] - mi) * (s_avg[s * NV + j] - mj);
        s_cov[e] = acc;
    }
    __syncthreads();
    for (int e = t; e < NV * NV; e += 256) {
        int i = e / NV, j = e % NV;
        float denom = sqrtf(s_cov[i * NV + i]) * sqrtf(s_cov[j * NV + j]);
        float c = fabsf(s_cov[e] / denom);
        if (isnan(c)) c = 0.f;
        if (i == j)  c = 0.f;
        ws[WS_CORR + e] = c;
    }
}

// ---------------------------------------------------------------------------
// Kernel 2: structure MLP 576->256->128->576, sigmoid, triu(k=1)
// ---------------------------------------------------------------------------
__global__ void __launch_bounds__(256)
k_struct(const float* __restrict__ sW1, const float* __restrict__ sb1,
         const float* __restrict__ sW2, const float* __restrict__ sb2,
         const float* __restrict__ sW3, const float* __restrict__ sb3,
         float* __restrict__ ws, float* __restrict__ out_adj) {
    __shared__ float s_c[NP];
    __shared__ float s_h1[NSH1];
    __shared__ float s_h2[NSH2];
    int t = threadIdx.x;

    for (int e = t; e < NP; e += 256) s_c[e] = ws[WS_CORR + e];
    __syncthreads();
    {
        float acc = sb1[t];
        for (int f = 0; f < NP; ++f) acc += s_c[f] * sW1[f * NSH1 + t];
        s_h1[t] = fmaxf(acc, 0.f);
    }
    __syncthreads();
    if (t < NSH2) {
        float acc = sb2[t];
        for (int f = 0; f < NSH1; ++f) acc += s_h1[f] * sW2[f * NSH2 + t];
        s_h2[t] = fmaxf(acc, 0.f);
    }
    __syncthreads();
    for (int e = t; e < NP; e += 256) {
        float acc = sb3[e];
        for (int f = 0; f < NSH2; ++f) acc += s_h2[f] * sW3[f * NP + e];
        float sg = 1.f / (1.f + __expf(-acc));
        int i = e / NV, j = e % NV;
        float a = (j > i) ? sg : 0.f;
        ws[WS_ADJ + e] = a;
        out_adj[e] = a;
    }
    __syncthreads();
    if (t < NV) {
        float hp = 0.f;
        for (int i = 0; i < NV; ++i)
            if (ws[WS_ADJ + i * NV + t] > 0.5f) hp = 1.f;
        ws[WS_HP + t] = hp;
    }
}

// ---------------------------------------------------------------------------
// Fragment-order LDS swizzles (16-bit operands, wave32), ISA 7.12.2:
//   A 16x32 (MxK):  lane = ((k&15)>>3)*16 + m,   elem = (k>>4)*8 + (k&7)
//   B 32x16 (KxN):  lane = ((k&15)>>3)*16 + n,   elem = (k>>4)*8 + (k&7)
// Each fragment tile = 512 halves; lane's 16 halves contiguous (32 B).
// ---------------------------------------------------------------------------
__device__ __forceinline__ int a_swz(int m, int k) {           // within one tile
    return (((k & 15) >> 3) * 16 + m) * 16 + ((k >> 4) * 8 + (k & 7));
}
__device__ __forceinline__ int b_swz(int k, int n) {           // within one tile
    return (((k & 15) >> 3) * 16 + n) * 16 + ((k >> 4) * 8 + (k & 7));
}

// ---------------------------------------------------------------------------
// Kernel 3: per-variable MLP via v_wmma_f32_16x16x32_f16
//   grid = (4096/64, 24), block = 128 threads (4 waves, 16 rows/wave)
// ---------------------------------------------------------------------------
#define TROWS 64
#define LDH2 (NH2 + 1)     // 65

__global__ void __launch_bounds__(128)
k_mlp(const float* __restrict__ data,
      const float* __restrict__ mW1, const float* __restrict__ mb1,
      const float* __restrict__ mW2, const float* __restrict__ mb2,
      const float* __restrict__ mW3, const float* __restrict__ mb3,
      const float* __restrict__ ws,  float* __restrict__ out_pred) {
    __shared__ __align__(32) _Float16 sAs[4 * 512];        // 4 row-tiles, A-order
    __shared__ __align__(32) _Float16 sW1s[8 * 512];       // 8 col-tiles, B-order
    __shared__ __align__(32) _Float16 sH1s[4 * 4 * 512];   // [wave][ktile], A-order
    __shared__ __align__(32) _Float16 sW2s[16 * 512];      // [c*4+kt], B-order
    __shared__ _Float16 sH2[TROWS * LDH2];                 // row-major for L3
    __shared__ float sB1[NH1];
    __shared__ float sB2[NH2];
    __shared__ float sW3[NH2];
    __shared__ float sMask[NV];

    const int v     = blockIdx.y;
    const int nBase = blockIdx.x * TROWS;
    const int t     = threadIdx.x;
    const int wave  = t >> 5;
    const int lane  = t & 31;

    // parent masks for this variable
    if (t < NV - 1) {
        int j = t + (t >= v ? 1 : 0);
        sMask[t] = (ws[WS_ADJ + j * NV + v] > 0.5f) ? 1.f : 0.f;
    }
    // W1 -> f16 LDS, B-fragment order (K padded 23->32)
    const float* W1v = mW1 + v * (NV - 1) * NH1;
    for (int e = t; e < 32 * NH1; e += 128) {
        int k = e >> 7, n = e & 127;
        float val = (k < NV - 1) ? W1v[k * NH1 + n] : 0.f;
        sW1s[(n >> 4) * 512 + b_swz(k, n & 15)] = (_Float16)val;
    }
    // W2 -> f16 LDS, B-fragment order (4 col-tiles x 4 k-tiles)
    const float* W2v = mW2 + v * NH1 * NH2;
    for (int e = t; e < NH1 * NH2; e += 128) {
        int k = e >> 6, n = e & 63;
        sW2s[((n >> 4) * 4 + (k >> 5)) * 512 + b_swz(k & 31, n & 15)] =
            (_Float16)W2v[e];
    }
    if (t < NH1) sB1[t] = mb1[v * NH1 + t];
    if (t < NH2) { sB2[t] = mb2[v * NH2 + t]; sW3[t] = mW3[v * NH2 + t]; }
    __syncthreads();

    // masked input tile -> A-fragment order
    for (int e = t; e < TROWS * 32; e += 128) {
        int r = e >> 5, f = e & 31;
        float val = 0.f;
        if (f < NV - 1) {
            int j = f + (f >= v ? 1 : 0);
            val = data[(nBase + r) * NV + j] * sMask[f];
        }
        sAs[(r >> 4) * 512 + a_swz(r & 15, f)] = (_Float16)val;
    }
    __syncthreads();

    const int n16 = lane & 15, mh = (lane >> 4) * 8;

    // Layer 1: (16x32)@(32x128): 8 WMMAs; write h1 straight into A-order
    {
        v16h a = *(const v16h*)(sAs + wave * 512 + lane * 16);
        #pragma unroll
        for (int c = 0; c < 8; ++c) {
            v8f acc = {};
            v16h b = *(const v16h*)(sW1s + c * 512 + lane * 16);
            acc = __builtin_amdgcn_wmma_f32_16x16x32_f16(
                      false, a, false, b, (short)0, acc, false, false);
            const int kt = c >> 1;
            _Float16* dst = sH1s + (wave * 4 + kt) * 512;
            #pragma unroll
            for (int i = 0; i < 8; ++i) {
                float x = fmaxf(acc[i] + sB1[c * 16 + n16], 0.f);
                int krel = (c & 1) * 16 + n16;          // k within 32-k tile
                dst[a_swz(mh + i, krel)] = (_Float16)x;
            }
        }
    }
    __syncthreads();   // (intra-wave dep only, but keep waves aligned)

    // Layer 2: (16x128)@(128x64): 4 col tiles x 4 K-steps = 16 WMMAs
    {
        v16h a2[4];
        #pragma unroll
        for (int kt = 0; kt < 4; ++kt)
            a2[kt] = *(const v16h*)(sH1s + (wave * 4 + kt) * 512 + lane * 16);
        #pragma unroll
        for (int c = 0; c < 4; ++c) {
            v8f acc = {};
            #pragma unroll
            for (int kt = 0; kt < 4; ++kt) {
                v16h b = *(const v16h*)(sW2s + (c * 4 + kt) * 512 + lane * 16);
                acc = __builtin_amdgcn_wmma_f32_16x16x32_f16(
                          false, a2[kt], false, b, (short)0, acc, false, false);
            }
            const int rw = wave * 16;
            #pragma unroll
            for (int i = 0; i < 8; ++i) {
                float x = fmaxf(acc[i] + sB2[c * 16 + n16], 0.f);
                sH2[(rw + mh + i) * LDH2 + c * 16 + n16] = (_Float16)x;
            }
        }
    }
    __syncthreads();

    // Layer 3: per-row dot over 64 + has_parent select
    if (t < TROWS) {
        float acc = mb3[v];
        #pragma unroll 8
        for (int g = 0; g < NH2; ++g)
            acc += (float)sH2[t * LDH2 + g] * sW3[g];
        int n = nBase + t;
        float dval = data[n * NV + v];
        out_pred[n * NV + v] = (ws[WS_HP + v] > 0.5f) ? acc : dval;
    }
}

// ---------------------------------------------------------------------------
// Kernel 4: pairwise interaction nets (K=2, N=1 -> VALU + block reduction)
// ---------------------------------------------------------------------------
__global__ void __launch_bounds__(256)
k_inter(const float* __restrict__ data,
        const float* __restrict__ iW1, const float* __restrict__ ib1,
        const float* __restrict__ iW2, const float* __restrict__ ib2,
        float* __restrict__ out_scores) {
    __shared__ float w0[NIH], w1[NIH], b1[NIH], w2[NIH];
    __shared__ float red[256];
    const int p = blockIdx.x;
    const int i = p / NV, j = p % NV;
    const int t = threadIdx.x;

    if (t < NIH) {
        w0[t] = iW1[p * 2 * NIH + t];
        w1[t] = iW1[p * 2 * NIH + NIH + t];
        b1[t] = ib1[p * NIH + t];
        w2[t] = iW2[p * NIH + t];
    }
    __syncthreads();
    float b2 = ib2[p];
    float sum = 0.f;
    for (int n = t; n < NROWS; n += 256) {
        float x0 = data[n * NV + i];
        float x1 = data[n * NV + j];
        float acc = b2;
        #pragma unroll
        for (int h = 0; h < NIH; ++h) {
            float a = fmaxf(x0 * w0[h] + x1 * w1[h] + b1[h], 0.f);
            acc += a * w2[h];
        }
        sum += 1.f / (1.f + __expf(-acc));
    }
    red[t] = sum;
    __syncthreads();
    for (int off = 128; off > 0; off >>= 1) {
        if (t < off) red[t] += red[t + off];
        __syncthreads();
    }
    if (t == 0) out_scores[p] = red[0] * (1.0f / NROWS);
}

// ---------------------------------------------------------------------------
extern "C" void kernel_launch(void* const* d_in, const int* in_sizes, int n_in,
                              void* d_out, int out_size, void* d_ws, size_t ws_size,
                              hipStream_t stream) {
    const float* data = (const float*)d_in[0];
    const float* sW1  = (const float*)d_in[1];
    const float* sb1  = (const float*)d_in[2];
    const float* sW2  = (const float*)d_in[3];
    const float* sb2  = (const float*)d_in[4];
    const float* sW3  = (const float*)d_in[5];
    const float* sb3  = (const float*)d_in[6];
    const float* mW1  = (const float*)d_in[7];
    const float* mb1  = (const float*)d_in[8];
    const float* mW2  = (const float*)d_in[9];
    const float* mb2  = (const float*)d_in[10];
    const float* mW3  = (const float*)d_in[11];
    const float* mb3  = (const float*)d_in[12];
    const float* iW1  = (const float*)d_in[13];
    const float* ib1  = (const float*)d_in[14];
    const float* iW2  = (const float*)d_in[15];
    const float* ib2  = (const float*)d_in[16];

    float* ws  = (float*)d_ws;
    float* out = (float*)d_out;
    float* out_adj    = out;                 // 576
    float* out_pred   = out + NP;            // 98304
    float* out_scores = out + NP + NROWS*NV; // 576

    k_stats <<<1, 256, 0, stream>>>(data, ws);
    k_struct<<<1, 256, 0, stream>>>(sW1, sb1, sW2, sb2, sW3, sb3, ws, out_adj);
    k_mlp   <<<dim3(NROWS / TROWS, NV), 128, 0, stream>>>(
                data, mW1, mb1, mW2, mb2, mW3, mb3, ws, out_pred);
    k_inter <<<NP, 256, 0, stream>>>(data, iW1, ib1, iW2, ib2, out_scores);
}
